// MultiHeadAttention_21663815041296
// MI455X (gfx1250) — matrix operand order repaired and checked by reading, not compiled, not run on hardware
//
#include <hip/hip_runtime.h>

// ---------------------------------------------------------------------------
// Multi-head attention for MI455X (gfx1250, wave32, WMMA).
//   S=4096, HIDDEN=768, 12 heads, D=64.
//   1) fp32 -> f16 conversion (workspace)
//   2) Q/K/V projections: v_wmma_f32_16x16x32_f16, copy-free double-buffered
//      (2x-unrolled pipeline), stored [head][seq][64], scale folded into Q
//   3) flash attention: S^T = K*Q^T so score C-layout == A-layout of P*V;
//      K staged by global_load_async_to_lds_b128 (ASYNCcnt) into ping-pong
//      buffers, V prefetched into registers and transpose-stored after compute
//   4) output projection -> fp32 d_out
// ---------------------------------------------------------------------------

#define HIDDEN 768
#define SEQ    4096
#define NHEAD  12
#define HDIM   64

typedef _Float16 h16 __attribute__((ext_vector_type(16)));
typedef _Float16 h8  __attribute__((ext_vector_type(8)));
typedef _Float16 h4  __attribute__((ext_vector_type(4)));
typedef float    f8  __attribute__((ext_vector_type(8)));
typedef int      v4i __attribute__((ext_vector_type(4)));

#if __has_builtin(__builtin_amdgcn_global_load_async_to_lds_b128) && \
    __has_builtin(__builtin_amdgcn_s_wait_asynccnt)
#define HAVE_ASYNC_LDS 1
typedef __attribute__((address_space(1))) v4i* gas_v4i_p;  // global
typedef __attribute__((address_space(3))) v4i* las_v4i_p;  // LDS
#else
#define HAVE_ASYNC_LDS 0
#endif

__device__ __forceinline__ h8 ld8(const _Float16* p) { return *(const h8*)p; }
__device__ __forceinline__ h16 cat8(h8 lo, h8 hi) {
  return __builtin_shufflevector(lo, hi, 0,1,2,3,4,5,6,7,8,9,10,11,12,13,14,15);
}
__device__ __forceinline__ f8 wmma_f16(h16 a, h16 b, f8 c) {
  return __builtin_amdgcn_wmma_f32_16x16x32_f16(false, a, false, b, (short)0, c,
                                                false, false);
}
__device__ __forceinline__ f8 f8zero() {
  f8 z = {0.f,0.f,0.f,0.f,0.f,0.f,0.f,0.f};
  return z;
}

// async global -> LDS (CDNA5 ASYNCcnt path) with synchronous fallback
__device__ __forceinline__ void stage16_async(const _Float16* g, _Float16* l) {
#if HAVE_ASYNC_LDS
  __builtin_amdgcn_global_load_async_to_lds_b128((gas_v4i_p)(const void*)g,
                                                 (las_v4i_p)(void*)l, 0, 0);
#else
  *(h8*)l = ld8(g);
#endif
}
__device__ __forceinline__ void wait_async() {
#if HAVE_ASYNC_LDS
  __builtin_amdgcn_s_wait_asynccnt(0);
#endif
}

// ---------------------------------------------------------------- convert ---
__global__ __launch_bounds__(256)
void cvt_f32_f16(const float* __restrict__ src, _Float16* __restrict__ dst, int n) {
  int i = (blockIdx.x * 256 + threadIdx.x) * 4;
  if (i < n) {
    float4 v = *(const float4*)(src + i);
    h4 o = { (_Float16)v.x, (_Float16)v.y, (_Float16)v.z, (_Float16)v.w };
    *(h4*)(dst + i) = o;
  }
}

// -------------------------------------------------- shared GEMM helpers -----
// One wave computes a 32(M) x 64(N) strip of X @ W^T.
__device__ __forceinline__ void gemm_load(const _Float16* xr0, const _Float16* xr1,
                                          const _Float16* wbase, int k0, int hb,
                                          h16& a0, h16& a1, h16 b[4]) {
  a0 = cat8(ld8(xr0 + k0 + 2 * hb), ld8(xr0 + k0 + 2 * hb + 8));
  a1 = cat8(ld8(xr1 + k0 + 2 * hb), ld8(xr1 + k0 + 2 * hb + 8));
  #pragma unroll
  for (int t = 0; t < 4; ++t) {
    const _Float16* wr = wbase + (size_t)t * 16 * HIDDEN + k0;
    b[t] = cat8(ld8(wr), ld8(wr + 8));
  }
}
__device__ __forceinline__ void gemm_mm(f8 acc[2][4], h16 a0, h16 a1, const h16 b[4]) {
  #pragma unroll
  for (int t = 0; t < 4; ++t) {
    acc[0][t] = wmma_f16(a0, b[t], acc[0][t]);
    acc[1][t] = wmma_f16(a1, b[t], acc[1][t]);
  }
}
__device__ __forceinline__ void gemm_pf(const _Float16* xr0, const _Float16* xr1,
                                        const _Float16* wbase, int k0) {
  __builtin_prefetch(xr0 + k0, 0, 3);          // near-scope prefetch
  __builtin_prefetch(xr1 + k0, 0, 3);
  #pragma unroll
  for (int t = 0; t < 4; ++t)
    __builtin_prefetch(wbase + (size_t)t * 16 * HIDDEN + k0, 0, 3);
}

// Copy-free software pipeline: 2x-unrolled ping-pong register buffers.
// Steps: s = k0/32, 24 total.  Preload s0->A; each loop iter computes two
// steps and loads the next two; peeled epilogue computes the last pair.
__device__ __forceinline__ void gemm_body(f8 acc[2][4], const _Float16* xr0,
                                          const _Float16* xr1, const _Float16* wbase,
                                          int hb) {
  h16 a0A, a1A, bA[4], a0B, a1B, bB[4];
  gemm_load(xr0, xr1, wbase, 0, hb, a0A, a1A, bA);
  #pragma unroll 1
  for (int k0 = 0; k0 < HIDDEN - 96; k0 += 64) {   // k0 = 0,64,...,640
    gemm_pf(xr0, xr1, wbase, k0 + 128);
    gemm_load(xr0, xr1, wbase, k0 + 32, hb, a0B, a1B, bB);
    gemm_mm(acc, a0A, a1A, bA);                    // step k0
    gemm_load(xr0, xr1, wbase, k0 + 64, hb, a0A, a1A, bA);
    gemm_mm(acc, a0B, a1B, bB);                    // step k0+32
  }
  // A holds step 704; load final step 736 and drain.
  gemm_load(xr0, xr1, wbase, HIDDEN - 32, hb, a0B, a1B, bB);
  gemm_mm(acc, a0A, a1A, bA);
  gemm_mm(acc, a0B, a1B, bB);
}

// ------------------------------------------------------ QKV projection ------
__global__ __launch_bounds__(256)
void proj_qkv(const _Float16* __restrict__ X, const _Float16* __restrict__ W,
              const float* __restrict__ bias, _Float16* __restrict__ dst,
              float oscale) {
  const int lane = threadIdx.x & 31;
  const int wave = threadIdx.x >> 5;
  const int l16  = lane & 15;
  const int hb   = (lane >> 4) ? 8 : 0;
  const int wt   = blockIdx.x * 8 + wave;       // 0..1535
  const int m0   = (wt / NHEAD) * 32;
  const int strip = wt % NHEAD;                 // == head
  const int n0   = strip * HDIM;

  f8 acc[2][4];
  #pragma unroll
  for (int mt = 0; mt < 2; ++mt)
    #pragma unroll
    for (int t = 0; t < 4; ++t) acc[mt][t] = f8zero();

  const _Float16* xr0 = X + (size_t)(m0 + l16) * HIDDEN;
  const _Float16* xr1 = X + (size_t)(m0 + 16 + l16) * HIDDEN;
  const _Float16* wbase = W + (size_t)(n0 + l16) * HIDDEN + hb * 2;
  gemm_body(acc, xr0, xr1, wbase, hb);

  #pragma unroll
  for (int t = 0; t < 4; ++t) {
    int d = t * 16 + l16;
    float bv = bias[n0 + d];
    #pragma unroll
    for (int mt = 0; mt < 2; ++mt)
      #pragma unroll
      for (int r = 0; r < 8; ++r) {
        int m = m0 + mt * 16 + hb + r;
        dst[((size_t)strip * SEQ + m) * HDIM + d] =
            (_Float16)((acc[mt][t][r] + bv) * oscale);
      }
  }
}

// --------------------------------------------------- output projection ------
__global__ __launch_bounds__(256)
void proj_out(const _Float16* __restrict__ X, const _Float16* __restrict__ W,
              const float* __restrict__ bias, float* __restrict__ out) {
  const int lane = threadIdx.x & 31;
  const int wave = threadIdx.x >> 5;
  const int l16  = lane & 15;
  const int hb   = (lane >> 4) ? 8 : 0;
  const int wt   = blockIdx.x * 8 + wave;
  const int m0   = (wt / NHEAD) * 32;
  const int n0   = (wt % NHEAD) * HDIM;

  f8 acc[2][4];
  #pragma unroll
  for (int mt = 0; mt < 2; ++mt)
    #pragma unroll
    for (int t = 0; t < 4; ++t) acc[mt][t] = f8zero();

  const _Float16* xr0 = X + (size_t)(m0 + l16) * HIDDEN;
  const _Float16* xr1 = X + (size_t)(m0 + 16 + l16) * HIDDEN;
  const _Float16* wbase = W + (size_t)(n0 + l16) * HIDDEN + hb * 2;
  gemm_body(acc, xr0, xr1, wbase, hb);

  #pragma unroll
  for (int t = 0; t < 4; ++t) {
    int n = n0 + t * 16 + l16;
    float bv = bias[n];
    #pragma unroll
    for (int mt = 0; mt < 2; ++mt)
      #pragma unroll
      for (int r = 0; r < 8; ++r)
        out[(size_t)(m0 + mt * 16 + hb + r) * HIDDEN + n] = acc[mt][t][r] + bv;
  }
}

// ---------------------------------------------------------- flash attn ------
// 8 waves x 16 queries = 128 queries per WG, one head per blockIdx.y.
// Ping-pong LDS: K[key][64] via async-to-LDS, V transposed [d][key] via
// register prefetch + ds_store after compute.
__global__ __launch_bounds__(256)
void flash_attn(const _Float16* __restrict__ Q, const _Float16* __restrict__ K,
                const _Float16* __restrict__ V, const float* __restrict__ mask,
                _Float16* __restrict__ ctx) {
  __shared__ __attribute__((aligned(16))) _Float16 Ks[2][32 * 80];
  __shared__ __attribute__((aligned(16))) _Float16 Vt[2][64 * 40];

  const int lane = threadIdx.x & 31;
  const int wave = threadIdx.x >> 5;
  const int l16  = lane & 15;
  const int hb   = (lane >> 4) ? 8 : 0;
  const int h    = blockIdx.y;
  const int q0   = (blockIdx.x * 8 + wave) * 16;

  // Q as B-matrix of S^T = K * Q^T: lane = query col, halves sweep d.
  const _Float16* qrow = Q + ((size_t)h * SEQ + q0 + l16) * HDIM + hb;
  h16 bq0 = cat8(ld8(qrow),      ld8(qrow + 16));   // d 0..31
  h16 bq1 = cat8(ld8(qrow + 32), ld8(qrow + 48));   // d 32..63

  f8 o[4];
  #pragma unroll
  for (int t = 0; t < 4; ++t) o[t] = f8zero();
  float mrun = -3.0e38f, lrun = 0.f;

  const int tkey = threadIdx.x >> 3;        // 0..31 staging row
  const int tdc  = (threadIdx.x & 7) * 8;   // 0..56 staging d-chunk
  const _Float16* Kg = K + ((size_t)h * SEQ + tkey) * HDIM + tdc;
  const _Float16* Vg = V + ((size_t)h * SEQ + tkey) * HDIM + tdc;

  // ---- prologue: stage block 0 into buffer 0
  stage16_async(Kg, &Ks[0][tkey * 80 + tdc]);
  {
    h8 vv = ld8(Vg);
    #pragma unroll
    for (int i = 0; i < 8; ++i) Vt[0][(tdc + i) * 40 + tkey] = vv[i];
  }

  const int nb = SEQ / 32;
  h8 vnext;
  for (int ib = 0; ib < nb; ++ib) {
    const int cur = ib & 1;
    const int kb  = ib * 32;
    wait_async();          // own async K slice done
    __syncthreads();       // + everyone's slices / V stores visible

    const bool more = (ib + 1) < nb;
    if (more) {            // issue next block's staging early (uniform branch)
      size_t nxt = (size_t)(kb + 32) * HDIM;
      stage16_async(Kg + nxt, &Ks[cur ^ 1][tkey * 80 + tdc]);
      vnext = ld8(Vg + nxt);
    }

    // ---- S^T tiles: rows = keys, cols = queries (scale already in Q)
    const _Float16* p00 = &Ks[cur][(0  + l16) * 80 + hb];
    const _Float16* p01 = &Ks[cur][(0  + l16) * 80 + 32 + hb];
    const _Float16* p10 = &Ks[cur][(16 + l16) * 80 + hb];
    const _Float16* p11 = &Ks[cur][(16 + l16) * 80 + 32 + hb];
    f8 s0 = wmma_f16(cat8(ld8(p01), ld8(p01 + 16)), bq1,
            wmma_f16(cat8(ld8(p00), ld8(p00 + 16)), bq0, f8zero()));
    f8 s1 = wmma_f16(cat8(ld8(p11), ld8(p11 + 16)), bq1,
            wmma_f16(cat8(ld8(p10), ld8(p10 + 16)), bq0, f8zero()));

    // ---- additive mask (per key = per C-row)
    float mv = mask[kb + lane];
    #pragma unroll
    for (int r = 0; r < 8; ++r) {
      s0[r] += __shfl(mv, hb + r, 32);
      s1[r] += __shfl(mv, 16 + hb + r, 32);
    }

    // ---- online softmax; lane pair (l, l+16) share query l16
    float mloc = -3.0e38f;
    #pragma unroll
    for (int r = 0; r < 8; ++r) mloc = fmaxf(mloc, fmaxf(s0[r], s1[r]));
    mloc = fmaxf(mloc, __shfl_xor(mloc, 16, 32));
    float mnew = fmaxf(mrun, mloc);
    float corr = __expf(mrun - mnew);

    float psum = 0.f;
    h16 pa;                 // P directly in A-layout (C of S^T == A of P)
    #pragma unroll
    for (int r = 0; r < 8; ++r) {
      float p0 = __expf(s0[r] - mnew);
      float p1 = __expf(s1[r] - mnew);
      psum += p0 + p1;
      pa[r]     = (_Float16)p0;   // keys hb..hb+7
      pa[r + 8] = (_Float16)p1;   // keys 16+hb..16+hb+7
    }
    psum += __shfl_xor(psum, 16, 32);
    lrun = lrun * corr + psum;
    mrun = mnew;

    // ---- O(16q x 64d) += P(16x32) * V(32x64)
    #pragma unroll
    for (int t = 0; t < 4; ++t) {
      const _Float16* pv = &Vt[cur][(t * 16 + l16) * 40 + hb];
      h16 bv = cat8(ld8(pv), ld8(pv + 16));
      #pragma unroll
      for (int r = 0; r < 8; ++r) o[t][r] *= __shfl(corr, hb + r, 32);
      o[t] = wmma_f16(pa, bv, o[t]);
    }

    // ---- V transpose-store for next block (load latency hidden by compute)
    if (more) {
      #pragma unroll
      for (int i = 0; i < 8; ++i) Vt[cur ^ 1][(tdc + i) * 40 + tkey] = vnext[i];
    }
  }

  float inv = 1.0f / lrun;
  #pragma unroll
  for (int t = 0; t < 4; ++t) {
    #pragma unroll
    for (int r = 0; r < 8; ++r) {
      float val = o[t][r] * __shfl(inv, hb + r, 32);
      ctx[(size_t)(q0 + hb + r) * HIDDEN + h * HDIM + t * 16 + l16] = (_Float16)val;
    }
  }
}

// ----------------------------------------------------------------- launch ---
extern "C" void kernel_launch(void* const* d_in, const int* in_sizes, int n_in,
                              void* d_out, int out_size, void* d_ws, size_t ws_size,
                              hipStream_t stream) {
  const float* x   = (const float*)d_in[0];
  const float* msk = (const float*)d_in[1];
  const float* qw  = (const float*)d_in[2];
  const float* qb  = (const float*)d_in[3];
  const float* kw  = (const float*)d_in[4];
  const float* kbb = (const float*)d_in[5];
  const float* vw  = (const float*)d_in[6];
  const float* vb  = (const float*)d_in[7];
  const float* ow  = (const float*)d_in[8];
  const float* ob  = (const float*)d_in[9];

  const size_t XN = (size_t)SEQ * HIDDEN;
  const size_t WN = (size_t)HIDDEN * HIDDEN;
  char* ws = (char*)d_ws;
  _Float16* Xh  = (_Float16*)ws; ws += XN * 2;
  _Float16* Wq  = (_Float16*)ws; ws += WN * 2;
  _Float16* Wk  = (_Float16*)ws; ws += WN * 2;
  _Float16* Wv  = (_Float16*)ws; ws += WN * 2;
  _Float16* Wo  = (_Float16*)ws; ws += WN * 2;
  _Float16* Qh  = (_Float16*)ws; ws += XN * 2;   // [12][4096][64]
  _Float16* Kh  = (_Float16*)ws; ws += XN * 2;
  _Float16* Vh  = (_Float16*)ws; ws += XN * 2;
  _Float16* Ctx = (_Float16*)ws; ws += XN * 2;   // ~36.2 MB total

  cvt_f32_f16<<<(int)((XN / 4 + 255) / 256), 256, 0, stream>>>(x,  Xh, (int)XN);
  cvt_f32_f16<<<(int)((WN / 4 + 255) / 256), 256, 0, stream>>>(qw, Wq, (int)WN);
  cvt_f32_f16<<<(int)((WN / 4 + 255) / 256), 256, 0, stream>>>(kw, Wk, (int)WN);
  cvt_f32_f16<<<(int)((WN / 4 + 255) / 256), 256, 0, stream>>>(vw, Wv, (int)WN);
  cvt_f32_f16<<<(int)((WN / 4 + 255) / 256), 256, 0, stream>>>(ow, Wo, (int)WN);

  const int projBlocks = (SEQ / 32) * NHEAD / 8;  // 192
  proj_qkv<<<projBlocks, 256, 0, stream>>>(Xh, Wq, qb,  Qh, 0.125f);
  proj_qkv<<<projBlocks, 256, 0, stream>>>(Xh, Wk, kbb, Kh, 1.0f);
  proj_qkv<<<projBlocks, 256, 0, stream>>>(Xh, Wv, vb,  Vh, 1.0f);

  flash_attn<<<dim3(SEQ / 128, NHEAD), 256, 0, stream>>>(Qh, Kh, Vh, msk, Ctx);

  proj_out<<<projBlocks, 256, 0, stream>>>(Ctx, Wo, ob, (float*)d_out);
}
